// DiffDimDotProductAttention_7533372637541
// MI455X (gfx1250) — compile-verified, matrix operand order checked by reading, not run on hardware
//
#include <hip/hip_runtime.h>
#include <math.h>

// Problem constants (match reference)
#define BB   8
#define LQ   2048
#define LK   2048
#define DQ   1024
#define DKEY 1024
#define DV   1024

typedef __attribute__((ext_vector_type(16))) __bf16 v16bf;
typedef __attribute__((ext_vector_type(8)))  __bf16 v8bf;
typedef __attribute__((ext_vector_type(4)))  __bf16 v4bf;
typedef __attribute__((ext_vector_type(8)))  float  v8f;

static __device__ __forceinline__ v16bf join8(v8bf lo, v8bf hi) {
  return __builtin_shufflevector(lo, hi, 0,1,2,3,4,5,6,7,8,9,10,11,12,13,14,15);
}

// ---------------------------------------------------------------------------
// Kernel 1: fp32 -> bf16 straight conversion (K matrix), vectorized 4-wide
// ---------------------------------------------------------------------------
__global__ __launch_bounds__(256) void k_cvt_bf16(const float4* __restrict__ src,
                                                  __bf16* __restrict__ dst, long n4) {
  long i = (long)blockIdx.x * blockDim.x + threadIdx.x;
  if (i >= n4) return;
  float4 v = src[i];
  v4bf o = { (__bf16)v.x, (__bf16)v.y, (__bf16)v.z, (__bf16)v.w };
  *(v4bf*)(dst + i * 4) = o;
}

// ---------------------------------------------------------------------------
// Kernel 2: tiled fp32 [rows][cols] -> bf16 [cols][rows] transpose (V, W_q)
// ---------------------------------------------------------------------------
__global__ __launch_bounds__(256) void k_transpose_bf16(const float* __restrict__ src,
                                                        __bf16* __restrict__ dst,
                                                        int rows, int cols) {
  __shared__ float tile[32][33];
  const int b = blockIdx.z;
  const float* s = src + (size_t)b * rows * cols;
  __bf16*      d = dst + (size_t)b * rows * cols;
  const int r0 = blockIdx.y * 32, c0 = blockIdx.x * 32;
  const int tx = threadIdx.x & 31, ty = threadIdx.x >> 5;   // 32 x 8
#pragma unroll
  for (int i = 0; i < 4; ++i) {
    int r = ty + i * 8;
    tile[r][tx] = s[(size_t)(r0 + r) * cols + c0 + tx];
  }
  __syncthreads();
#pragma unroll
  for (int i = 0; i < 4; ++i) {
    int c = ty + i * 8;
    d[(size_t)(c0 + c) * rows + r0 + tx] = (__bf16)tile[tx][c];
  }
}

// ---------------------------------------------------------------------------
// Kernel 3: q projection  qp[b,q,n] = sum_k queries[b,q,k] * Wq[k,n]
//   A: queries fp32 -> bf16 staged in LDS (16 rows x 1024)
//   B: Wt = Wq^T bf16 [DKEY][DQ]  (contiguous along reduction dim)
//   grid.x = B*LQ/16, 256 threads = 8 waves, wave w owns output cols [128w,128w+128)
// ---------------------------------------------------------------------------
__global__ __launch_bounds__(256) void k_qproj(const float* __restrict__ Q,
                                               const __bf16* __restrict__ Wt,
                                               __bf16* __restrict__ qp) {
  const int wg = blockIdx.x;
  const int b  = wg / (LQ / 16);
  const int q0 = (wg % (LQ / 16)) * 16;

  __shared__ __bf16 As[16 * DQ];          // 32 KB

  const float* qsrc = Q + ((size_t)b * LQ + q0) * DQ;
  for (int idx = threadIdx.x * 4; idx < 16 * DQ; idx += 256 * 4) {
    float4 v = *(const float4*)(qsrc + idx);
    v4bf o = { (__bf16)v.x, (__bf16)v.y, (__bf16)v.z, (__bf16)v.w };
    *(v4bf*)(As + idx) = o;
  }
  __syncthreads();

  const int w = threadIdx.x >> 5, l = threadIdx.x & 31;
  const int m = l & 15;                    // A row / B col / C col
  const int k0 = (l >> 4) * 8;            // K offset per half-wave
  const int rowblk = 8 * (l >> 4);        // C row base

  v8f acc[8];
#pragma unroll
  for (int f = 0; f < 8; ++f) acc[f] = (v8f){0.f,0.f,0.f,0.f,0.f,0.f,0.f,0.f};

  for (int kc = 0; kc < DQ; kc += 32) {
    v16bf a = join8(*(const v8bf*)&As[m * DQ + kc + k0],
                    *(const v8bf*)&As[m * DQ + kc + k0 + 16]);
#pragma unroll
    for (int f = 0; f < 8; ++f) {
      const __bf16* bp = Wt + (size_t)(w * 128 + f * 16 + m) * DQ + kc;
      v16bf bb = join8(*(const v8bf*)(bp + k0), *(const v8bf*)(bp + k0 + 16));
      acc[f] = __builtin_amdgcn_wmma_f32_16x16x32_bf16(false, a, false, bb,
                                                       (short)0, acc[f], false, false);
    }
  }

  __bf16* dst = qp + ((size_t)b * LQ + q0) * DKEY;
#pragma unroll
  for (int f = 0; f < 8; ++f)
#pragma unroll
    for (int r = 0; r < 8; ++r)
      dst[(size_t)(rowblk + r) * DKEY + w * 128 + f * 16 + m] = (__bf16)acc[f][r];
}

// ---------------------------------------------------------------------------
// Kernel 4: flash attention with online softmax + length mask.
//   Workgroup: 16 query rows, 8 waves. Key tiles of 128 (wave w owns 16 keys
//   for S, and DV slice [128w,128w+128) for O). All GEMMs via bf16 WMMA.
// ---------------------------------------------------------------------------
__global__ __launch_bounds__(256) void k_flash(const __bf16* __restrict__ qp,
                                               const __bf16* __restrict__ Kb,
                                               const __bf16* __restrict__ Vt,
                                               const int* __restrict__ vlen,
                                               float* __restrict__ out) {
  const int wg = blockIdx.x;
  const int b  = wg / (LQ / 16);
  const int q0 = (wg % (LQ / 16)) * 16;
  const int vl = vlen[b];

  __shared__ __bf16 Qs[16 * DKEY];        // 32 KB projected q tile (bf16)
  __shared__ float  Ss[16][128];          //  8 KB scores tile
  __shared__ __bf16 Ps[16][128];          //  4 KB probs tile (bf16)
  __shared__ float  red[16][16];
  __shared__ float  mrow[16], lrow[16], arow[16];

  const __bf16* qsrc = qp + ((size_t)b * LQ + q0) * DKEY;
  for (int idx = threadIdx.x * 8; idx < 16 * DKEY; idx += 256 * 8)
    *(v8bf*)&Qs[idx] = *(const v8bf*)(qsrc + idx);
  if (threadIdx.x < 16) { mrow[threadIdx.x] = -INFINITY; lrow[threadIdx.x] = 0.f; }

  const int w = threadIdx.x >> 5, l = threadIdx.x & 31;
  const int m = l & 15;
  const int k0 = (l >> 4) * 8;
  const int rowblk = 8 * (l >> 4);
  const int sr = threadIdx.x >> 4;        // softmax row (0..15)
  const int sj = threadIdx.x & 15;        // 16 threads/row, 8 cols each

  const __bf16* Kbase = Kb + (size_t)b * LK * DKEY;
  const __bf16* Vbase = Vt + (size_t)b * DV * LK;
  const float scale = 0.03125f;           // 1/sqrt(1024)

  v8f o[8];
#pragma unroll
  for (int f = 0; f < 8; ++f) o[f] = (v8f){0.f,0.f,0.f,0.f,0.f,0.f,0.f,0.f};

  __syncthreads();

  for (int kt = 0; kt < LK; kt += 128) {
    // ---- S tile: wave w computes 16x16 for keys [kt+16w, kt+16w+16)
    v8f s = (v8f){0.f,0.f,0.f,0.f,0.f,0.f,0.f,0.f};
    const __bf16* krow = Kbase + (size_t)(kt + 16 * w + m) * DKEY;  // B col = m
    for (int kc = 0; kc < DKEY; kc += 32) {
      v16bf a  = join8(*(const v8bf*)&Qs[m * DKEY + kc + k0],
                       *(const v8bf*)&Qs[m * DKEY + kc + k0 + 16]);
      v16bf bb = join8(*(const v8bf*)(krow + kc + k0),
                       *(const v8bf*)(krow + kc + k0 + 16));
      s = __builtin_amdgcn_wmma_f32_16x16x32_bf16(false, a, false, bb,
                                                  (short)0, s, false, false);
    }
    // scale + length mask, spill to LDS
    const bool valid = (kt + 16 * w + m) < vl;
#pragma unroll
    for (int r = 0; r < 8; ++r)
      Ss[rowblk + r][16 * w + m] = valid ? s[r] * scale : -1.0e6f;
    __syncthreads();

    // ---- online softmax over the 128-wide tile
    float lm = -INFINITY;
#pragma unroll
    for (int c = 0; c < 8; ++c) lm = fmaxf(lm, Ss[sr][sj * 8 + c]);
    red[sr][sj] = lm;
    __syncthreads();
    if (sj == 0) {
      float rm = red[sr][0];
#pragma unroll
      for (int c = 1; c < 16; ++c) rm = fmaxf(rm, red[sr][c]);
      float mold = mrow[sr];
      float mnew = fmaxf(mold, rm);
      mrow[sr] = mnew;
      arow[sr] = __expf(mold - mnew);
    }
    __syncthreads();
    const float mnew = mrow[sr];
    float psum = 0.f;
#pragma unroll
    for (int c = 0; c < 8; ++c) {
      float e = __expf(Ss[sr][sj * 8 + c] - mnew);
      Ps[sr][sj * 8 + c] = (__bf16)e;
      psum += e;
    }
    red[sr][sj] = psum;
    __syncthreads();
    if (sj == 0) {
      float tot = 0.f;
#pragma unroll
      for (int c = 0; c < 16; ++c) tot += red[sr][c];
      lrow[sr] = arow[sr] * lrow[sr] + tot;
    }
    __syncthreads();

    // ---- rescale O by alpha, then O += P @ V  (wave w: DV cols 128w..128w+127)
    float al[8];
#pragma unroll
    for (int r = 0; r < 8; ++r) al[r] = arow[rowblk + r];
#pragma unroll
    for (int f = 0; f < 8; ++f)
#pragma unroll
      for (int r = 0; r < 8; ++r) o[f][r] *= al[r];

#pragma unroll
    for (int kc = 0; kc < 128; kc += 32) {
      v16bf a = join8(*(const v8bf*)&Ps[m][kc + k0],
                      *(const v8bf*)&Ps[m][kc + k0 + 16]);
#pragma unroll
      for (int f = 0; f < 8; ++f) {
        const __bf16* vr = Vbase + (size_t)(128 * w + 16 * f + m) * LK + kt + kc;
        v16bf bb = join8(*(const v8bf*)(vr + k0), *(const v8bf*)(vr + k0 + 16));
        o[f] = __builtin_amdgcn_wmma_f32_16x16x32_bf16(false, a, false, bb,
                                                       (short)0, o[f], false, false);
      }
    }
    __syncthreads();
  }

  // ---- epilogue: normalize by l and store fp32
  float invl[8];
#pragma unroll
  for (int r = 0; r < 8; ++r) invl[r] = 1.0f / lrow[rowblk + r];
  float* dst = out + ((size_t)b * LQ + q0) * DV;
#pragma unroll
  for (int f = 0; f < 8; ++f)
#pragma unroll
    for (int r = 0; r < 8; ++r)
      dst[(size_t)(rowblk + r) * DV + 128 * w + 16 * f + m] = o[f][r] * invl[r];
}

// ---------------------------------------------------------------------------
extern "C" void kernel_launch(void* const* d_in, const int* in_sizes, int n_in,
                              void* d_out, int out_size, void* d_ws, size_t ws_size,
                              hipStream_t stream) {
  const float* queries = (const float*)d_in[0];
  const float* keys    = (const float*)d_in[1];
  const float* values  = (const float*)d_in[2];
  const int*   vlens   = (const int*)  d_in[3];
  const float* Wq      = (const float*)d_in[4];
  float*       out     = (float*)d_out;

  // Workspace layout (bf16): K | V^T | Wq^T | projected q   (~98 MB)
  const size_t nK = (size_t)BB * LK * DKEY;
  const size_t nV = (size_t)BB * LK * DV;
  const size_t nW = (size_t)DQ * DKEY;
  __bf16* Kb = (__bf16*)d_ws;
  __bf16* Vt = Kb + nK;
  __bf16* Wt = Vt + nV;
  __bf16* qp = Wt + nW;

  // 1) K -> bf16 (row-major, contiguous along DKEY)
  {
    long n4 = (long)nK / 4;
    long blocks = (n4 + 255) / 256;
    k_cvt_bf16<<<(unsigned)blocks, 256, 0, stream>>>((const float4*)keys, Kb, n4);
  }
  // 2) V -> bf16 transposed [B][DV][LK]
  {
    dim3 g(DV / 32, LK / 32, BB);
    k_transpose_bf16<<<g, 256, 0, stream>>>(values, Vt, LK, DV);
  }
  // 3) Wq -> bf16 transposed [DKEY][DQ]
  {
    dim3 g(DKEY / 32, DQ / 32, 1);
    k_transpose_bf16<<<g, 256, 0, stream>>>(Wq, Wt, DQ, DKEY);
  }
  // 4) q projection (bf16 WMMA)
  k_qproj<<<BB * LQ / 16, 256, 0, stream>>>(queries, Wt, qp);
  // 5) fused flash attention (bf16 WMMA, online softmax, length mask)
  k_flash<<<BB * LQ / 16, 256, 0, stream>>>(qp, Kb, Vt, vlens, out);
}
